// CrossAttention_38783554683574
// MI455X (gfx1250) — matrix operand order repaired and checked by reading, not compiled, not run on hardware
//
#include <hip/hip_runtime.h>

typedef __attribute__((ext_vector_type(16))) _Float16 v16h;
typedef __attribute__((ext_vector_type(8)))  _Float16 v8h;
typedef __attribute__((ext_vector_type(8)))  float    v8f;

#define BATCH 4
#define CCH   64
#define NPIX  4096   // 64*64

// ---------------------------------------------------------------------------
// Phase 1: all 1x1-conv projections, fp32 in -> f16 out, WMMA-friendly layouts
//   qT  : [b][n][16]   (rows 0..7 = Wq1*x1, rows 8..15 = Wq2*x2)  channel-contig
//   kT  : [b][n][16]   channel-contiguous (transposed for B-operand build)
//   v   : [b][c][n]    n-contiguous (P*V contracts over n)
// ---------------------------------------------------------------------------
__global__ __launch_bounds__(256)
void proj_kernel(const float* __restrict__ x1, const float* __restrict__ x2,
                 const float* __restrict__ Wq1, const float* __restrict__ bq1,
                 const float* __restrict__ Wq2, const float* __restrict__ bq2,
                 const float* __restrict__ Wk1, const float* __restrict__ bk1,
                 const float* __restrict__ Wk2, const float* __restrict__ bk2,
                 const float* __restrict__ Wv1, const float* __restrict__ bv1,
                 const float* __restrict__ Wv2, const float* __restrict__ bv2,
                 _Float16* __restrict__ qT,  _Float16* __restrict__ k1T,
                 _Float16* __restrict__ k2T, _Float16* __restrict__ v1h,
                 _Float16* __restrict__ v2h)
{
    const int n = blockIdx.x * blockDim.x + threadIdx.x;   // spatial position
    const int o = blockIdx.y;                              // global output channel 0..175
    const int b = blockIdx.z;

    const float* W; const float* bias; const float* x;
    int oo; _Float16* dst; size_t didx;

    if (o < 8)        { W=Wq1; bias=bq1; x=x1; oo=o;     dst=qT;  didx=((size_t)b*NPIX+n)*16 + o;        }
    else if (o < 16)  { W=Wq2; bias=bq2; x=x2; oo=o-8;   dst=qT;  didx=((size_t)b*NPIX+n)*16 + o;        }
    else if (o < 32)  { W=Wk1; bias=bk1; x=x1; oo=o-16;  dst=k1T; didx=((size_t)b*NPIX+n)*16 + oo;       }
    else if (o < 48)  { W=Wk2; bias=bk2; x=x2; oo=o-32;  dst=k2T; didx=((size_t)b*NPIX+n)*16 + oo;       }
    else if (o < 112) { W=Wv1; bias=bv1; x=x1; oo=o-48;  dst=v1h; didx=((size_t)b*CCH+oo)*NPIX + n;      }
    else              { W=Wv2; bias=bv2; x=x2; oo=o-112; dst=v2h; didx=((size_t)b*CCH+oo)*NPIX + n;      }

    const float* wr = W + oo * CCH;                    // uniform per block -> scalar loads
    const float* xp = x + (size_t)b * CCH * NPIX + n;  // coalesced across threads
    float acc = bias[oo];
    #pragma unroll
    for (int c = 0; c < CCH; ++c)
        acc = fmaf(wr[c], xp[(size_t)c * NPIX], acc);
    dst[didx] = (_Float16)acc;
}

// ---------------------------------------------------------------------------
// Phase 2: fused flash attention + residual.
// Grid: (N/64 m-blocks, 2 attends, B). Block = 128 threads = 4 waves.
// 64-key chunks: 4 QK WMMAs + 8 PV WMMAs per iteration. Software-pipelined:
// V loads issued at body top (covered by QK+softmax), K operands preloaded
// one chunk ahead in persistent registers.
// ---------------------------------------------------------------------------
__global__ __launch_bounds__(128)
void flash_kernel(const float* __restrict__ x1, const float* __restrict__ x2,
                  const _Float16* __restrict__ qT,
                  const _Float16* __restrict__ k1T, const _Float16* __restrict__ k2T,
                  const _Float16* __restrict__ v1h, const _Float16* __restrict__ v2h,
                  float* __restrict__ out)
{
    // wave-private P-transpose staging; 72-half (144B) row stride keeps the
    // 16 row-reads on distinct bank groups while preserving 16B alignment.
    __shared__ __align__(16) _Float16 plds[4][16][72];

    const int wave    = threadIdx.x >> 5;
    const int lane    = threadIdx.x & 31;
    const int lanem   = lane & 15;
    const int halfSel = lane >> 4;
    const int klo     = halfSel * 8;

    const int b   = blockIdx.z;
    const int att = blockIdx.y;
    const int m0  = blockIdx.x * 64 + wave * 16;

    const _Float16* kT = att ? k2T : k1T;
    const _Float16* vh = att ? v2h : v1h;
    const float* xb = (att ? x2 : x1) + (size_t)b * CCH * NPIX;
    float* ob = out + (size_t)att * BATCH * CCH * NPIX + (size_t)b * CCH * NPIX;

    // ---- Q A-operand (16x32 f16 A-layout; real K=16, upper half zero).
    // Because A's K=16..31 are zero, the B operand's K=16..31 lanes may hold
    // arbitrary finite data -- no masking needed on the K loads.
    v16h aq;
    {
        const _Float16* qrow = qT + ((size_t)b * NPIX + m0 + lanem) * 16;
        v8h qlo = *(const v8h*)qrow;
        v8h qhi = *(const v8h*)(qrow + 8);
        #pragma unroll
        for (int i = 0; i < 8; ++i) { aq[i] = halfSel ? (_Float16)0.0f : qlo[i]; aq[i + 8] = halfSel ? (_Float16)0.0f : qhi[i]; }
    }

    v8f acc0 = {}, acc1 = {}, acc2 = {}, acc3 = {};
    float rmax[8], rsum[8];           // rsum is a PER-LANE partial sum
    #pragma unroll
    for (int r = 0; r < 8; ++r) { rmax[r] = -3.0e38f; rsum[r] = 0.0f; }

    const _Float16* kbb = kT + ((size_t)b * NPIX + lanem) * 16;
    const _Float16* vbb = vh + ((size_t)b * CCH + lanem) * NPIX + halfSel * 16;

    // ---- K pipeline: preload chunk 0 into persistent registers ----
    v16h bk0 = *(const v16h*)(kbb);
    v16h bk1 = *(const v16h*)(kbb + 16 * 16);
    v16h bk2 = *(const v16h*)(kbb + 32 * 16);
    v16h bk3 = *(const v16h*)(kbb + 48 * 16);

    for (int n0 = 0; n0 < NPIX; n0 += 64) {
        // ---- V loads for the CURRENT chunk, issued first so the whole
        //      QK + softmax section hides their latency ----
        const _Float16* vb = vbb + n0;
        v16h bv00 = *(const v16h*)(vb);
        v16h bv01 = *(const v16h*)(vb + 32);
        v16h bv10 = *(const v16h*)(vb + (size_t)16 * NPIX);
        v16h bv11 = *(const v16h*)(vb + (size_t)16 * NPIX + 32);
        v16h bv20 = *(const v16h*)(vb + (size_t)32 * NPIX);
        v16h bv21 = *(const v16h*)(vb + (size_t)32 * NPIX + 32);
        v16h bv30 = *(const v16h*)(vb + (size_t)48 * NPIX);
        v16h bv31 = *(const v16h*)(vb + (size_t)48 * NPIX + 32);

        if (n0 + 64 < NPIX) {   // speculative prefetch two chunks ahead
            __builtin_prefetch(vbb + n0 + 128, 0, 1);
            __builtin_prefetch(vbb + (size_t)32 * NPIX + n0 + 128, 0, 1);
        }

        // ---- QK^T on preloaded K operands ----
        v8f zc = {};
        v8f s0 = __builtin_amdgcn_wmma_f32_16x16x32_f16(false, aq, false, bk0, (short)0, zc, false, false);
        v8f s1 = __builtin_amdgcn_wmma_f32_16x16x32_f16(false, aq, false, bk1, (short)0, zc, false, false);
        v8f s2 = __builtin_amdgcn_wmma_f32_16x16x32_f16(false, aq, false, bk2, (short)0, zc, false, false);
        v8f s3 = __builtin_amdgcn_wmma_f32_16x16x32_f16(false, aq, false, bk3, (short)0, zc, false, false);

        // ---- preload NEXT chunk's K (clamped address: branch-free, in-bounds;
        //      last iteration just reloads the final chunk) ----
        {
            int nn = n0 + 64 < NPIX ? n0 + 64 : n0;
            const _Float16* kb = kbb + (size_t)nn * 16;
            bk0 = *(const v16h*)(kb);
            bk1 = *(const v16h*)(kb + 16 * 16);
            bk2 = *(const v16h*)(kb + 32 * 16);
            bk3 = *(const v16h*)(kb + 48 * 16);
        }

        // ---- online softmax. D-layout: reg r, lane L -> row r+8*(L>=16), col L%16.
        // xor masks 1,2,4,8 reduce each 16-lane half (i.e. each row) independently.
        float scale[8];
        #pragma unroll
        for (int r = 0; r < 8; ++r) {
            float mx = fmaxf(fmaxf(s0[r], s1[r]), fmaxf(s2[r], s3[r]));
            mx = fmaxf(mx, __shfl_xor(mx, 1));
            mx = fmaxf(mx, __shfl_xor(mx, 2));
            mx = fmaxf(mx, __shfl_xor(mx, 4));
            mx = fmaxf(mx, __shfl_xor(mx, 8));
            float nm = fmaxf(rmax[r], mx);
            float sc = __expf(rmax[r] - nm);
            rmax[r] = nm;
            float p0 = __expf(s0[r] - nm);
            float p1 = __expf(s1[r] - nm);
            float p2 = __expf(s2[r] - nm);
            float p3 = __expf(s3[r] - nm);
            rsum[r] = rsum[r] * sc + ((p0 + p1) + (p2 + p3));   // per-lane partial
            scale[r] = sc;
            _Float16* prow = &plds[wave][r + halfSel * 8][0];
            prow[lanem]      = (_Float16)p0;
            prow[lanem + 16] = (_Float16)p1;
            prow[lanem + 32] = (_Float16)p2;
            prow[lanem + 48] = (_Float16)p3;
        }
        #pragma unroll
        for (int r = 0; r < 8; ++r) {
            acc0[r] *= scale[r]; acc1[r] *= scale[r];
            acc2[r] *= scale[r]; acc3[r] *= scale[r];
        }

        // wave-private LDS round trip (DS ops in-order per wave; wait to be safe)
        asm volatile("s_wait_dscnt 0" ::: "memory");

        // ---- read P back as two A-layout operands (16 rows x K=32 each) ----
        v16h ap0, ap1;
        {
            const _Float16* pr = &plds[wave][lanem][0];
            v8h a0 = *(const v8h*)(pr + 2 * klo);
            v8h a1 = *(const v8h*)(pr + 8 + 2 * klo);
            v8h a2 = *(const v8h*)(pr + 32 + 2 * klo);
            v8h a3 = *(const v8h*)(pr + 40 + 2 * klo);
            #pragma unroll
            for (int i = 0; i < 8; ++i) {
                ap0[i] = a0[i]; ap0[i + 8] = a1[i];
                ap1[i] = a2[i]; ap1[i + 8] = a3[i];
            }
        }

        // ---- P*V: 4 column tiles of V (Cv=64) x 2 K-sub-chunks.
        // B-layout: lane col = c; lanes0-15 K(n)=n0..n0+15, lanes16-31 n0+16..n0+31.
        acc0 = __builtin_amdgcn_wmma_f32_16x16x32_f16(false, ap0, false, bv00, (short)0, acc0, false, false);
        acc0 = __builtin_amdgcn_wmma_f32_16x16x32_f16(false, ap1, false, bv01, (short)0, acc0, false, false);
        acc1 = __builtin_amdgcn_wmma_f32_16x16x32_f16(false, ap0, false, bv10, (short)0, acc1, false, false);
        acc1 = __builtin_amdgcn_wmma_f32_16x16x32_f16(false, ap1, false, bv11, (short)0, acc1, false, false);
        acc2 = __builtin_amdgcn_wmma_f32_16x16x32_f16(false, ap0, false, bv20, (short)0, acc2, false, false);
        acc2 = __builtin_amdgcn_wmma_f32_16x16x32_f16(false, ap1, false, bv21, (short)0, acc2, false, false);
        acc3 = __builtin_amdgcn_wmma_f32_16x16x32_f16(false, ap0, false, bv30, (short)0, acc3, false, false);
        acc3 = __builtin_amdgcn_wmma_f32_16x16x32_f16(false, ap1, false, bv31, (short)0, acc3, false, false);
    }

    // ---- deferred cross-lane reduction of the per-lane row sums (once) ----
    #pragma unroll
    for (int r = 0; r < 8; ++r) {
        float s = rsum[r];
        s += __shfl_xor(s, 1);
        s += __shfl_xor(s, 2);
        s += __shfl_xor(s, 4);
        s += __shfl_xor(s, 8);
        rsum[r] = s;
    }

    // ---- epilogue: normalize, add residual, store.
    // D-layout: reg r, lane L -> row m = m0 + r + 8*halfSel, col c = tile*16 + L%16
    #pragma unroll
    for (int r = 0; r < 8; ++r) {
        const float inv = 1.0f / rsum[r];
        const int m = m0 + r + halfSel * 8;
        const size_t c0 = (size_t)lanem;
        float o0 = acc0[r] * inv + xb[(c0     ) * NPIX + m];
        float o1 = acc1[r] * inv + xb[(c0 + 16) * NPIX + m];
        float o2 = acc2[r] * inv + xb[(c0 + 32) * NPIX + m];
        float o3 = acc3[r] * inv + xb[(c0 + 48) * NPIX + m];
        ob[(c0     ) * NPIX + m] = o0;
        ob[(c0 + 16) * NPIX + m] = o1;
        ob[(c0 + 32) * NPIX + m] = o2;
        ob[(c0 + 48) * NPIX + m] = o3;
    }
}

// ---------------------------------------------------------------------------
extern "C" void kernel_launch(void* const* d_in, const int* in_sizes, int n_in,
                              void* d_out, int out_size, void* d_ws, size_t ws_size,
                              hipStream_t stream)
{
    (void)in_sizes; (void)n_in; (void)out_size; (void)ws_size;

    const float* x1  = (const float*)d_in[0];
    const float* x2  = (const float*)d_in[1];
    const float* Wq1 = (const float*)d_in[2];  const float* bq1 = (const float*)d_in[3];
    const float* Wq2 = (const float*)d_in[4];  const float* bq2 = (const float*)d_in[5];
    const float* Wk1 = (const float*)d_in[6];  const float* bk1 = (const float*)d_in[7];
    const float* Wk2 = (const float*)d_in[8];  const float* bk2 = (const float*)d_in[9];
    const float* Wv1 = (const float*)d_in[10]; const float* bv1 = (const float*)d_in[11];
    const float* Wv2 = (const float*)d_in[12]; const float* bv2 = (const float*)d_in[13];

    // workspace: f16 staging (~5.8 MB)
    _Float16* qT  = (_Float16*)d_ws;
    _Float16* k1T = qT  + (size_t)BATCH * 16  * NPIX;
    _Float16* k2T = k1T + (size_t)BATCH * 16  * NPIX;
    _Float16* v1h = k2T + (size_t)BATCH * 16  * NPIX;
    _Float16* v2h = v1h + (size_t)BATCH * CCH * NPIX;

    dim3 g1(NPIX / 256, 176, BATCH);
    proj_kernel<<<g1, 256, 0, stream>>>(x1, x2, Wq1, bq1, Wq2, bq2, Wk1, bk1,
                                        Wk2, bk2, Wv1, bv1, Wv2, bv2,
                                        qT, k1T, k2T, v1h, v2h);

    dim3 g2(NPIX / 64, 2, BATCH);
    flash_kernel<<<g2, 128, 0, stream>>>(x1, x2, qT, k1T, k2T, v1h, v2h,
                                         (float*)d_out);
}
